// RPEMHALayer_43954695307294
// MI455X (gfx1250) — compile-verified, matrix operand order checked by reading, not run on hardware
//
#include <hip/hip_runtime.h>
#include <hip/hip_bf16.h>

typedef __bf16 bf16_t;
typedef __attribute__((ext_vector_type(16))) __bf16 v16bf;
typedef __attribute__((ext_vector_type(8)))  __bf16 v8bf;
typedef __attribute__((ext_vector_type(8)))  float  v8f;

#define B_   4
#define LQ_  256
#define LK_  512
#define DIN_ 128
#define H_   8
#define NB_  32
#define NROWS_R (B_*LQ_*NB_)   // 32768 gathered rpe rows

#ifndef __has_builtin
#define __has_builtin(x) 0
#endif
#if __has_builtin(__builtin_amdgcn_tensor_load_to_lds) && __has_builtin(__builtin_amdgcn_s_wait_tensorcnt)
#define HAS_TDM 1
#else
#define HAS_TDM 0
#endif

#if HAS_TDM
typedef unsigned int u32x4 __attribute__((ext_vector_type(4)));
typedef int          i32x4 __attribute__((ext_vector_type(4)));
typedef int          i32x8 __attribute__((ext_vector_type(8)));

// TDM: DMA one 128x128 bf16 (2-byte) tile global -> LDS. D# packed per ISA 08_async_tensor.
__device__ __forceinline__ void tdm_load_w_128x128(const bf16_t* gsrc, unsigned lds_off) {
  unsigned long long ga = (unsigned long long)gsrc;
  u32x4 g0;
  g0[0] = 1u;                                           // count=1, is_restore=0, gather=0
  g0[1] = lds_off;                                      // lds_addr (bytes)
  g0[2] = (unsigned)(ga & 0xffffffffu);                 // global_addr[31:0]
  g0[3] = (unsigned)((ga >> 32) & 0x01ffffffu)          // global_addr[56:32]
          | (2u << 30);                                 // type=2 ("image")
  i32x8 g1;
  g1[0] = 0x00010000;                                   // wg_mask=0, data_size=1 (2 bytes)
  g1[1] = (int)(128u << 16);                            // tensor_dim0=128 (bits 79:48, lo16)
  g1[2] = (int)(128u << 16);                            // tensor_dim1=128 (bits 111:80, lo16)
  g1[3] = (int)(128u << 16);                            // tile_dim0=128  (bits 127:112)
  g1[4] = 128;                                          // tile_dim1=128  (bits 143:128)
  g1[5] = 128;                                          // tensor_dim0_stride=128 (bits 207:160)
  g1[6] = 0;
  g1[7] = 0;
  i32x4 z4 = {0, 0, 0, 0};
#if __clang_major__ >= 23
  i32x8 z8 = {0, 0, 0, 0, 0, 0, 0, 0};
  __builtin_amdgcn_tensor_load_to_lds(g0, g1, z4, z4, z8, 0);
#else
  __builtin_amdgcn_tensor_load_to_lds(g0, g1, z4, z4, 0);
#endif
}
#endif  // HAS_TDM

// ---------------- wave helpers ----------------
__device__ __forceinline__ float wave_sum32(float v) {
#pragma unroll
  for (int off = 16; off > 0; off >>= 1) v += __shfl_xor(v, off, 32);
  return v;
}

// ---------------- weight prep: fp32 [K][N] -> bf16 [N][K] ----------------
__global__ void prep_w_kernel(const float* __restrict__ W, bf16_t* __restrict__ Wt) {
  int e = blockIdx.x * 256 + threadIdx.x;   // 0..16383
  int k = e >> 7, n = e & 127;
  Wt[(size_t)n * 128 + k] = (bf16_t)W[(size_t)k * 128 + n];
}

// ---------------- layernorm over 128-wide rows, one wave per row ----------------
__device__ __forceinline__ void ln_row_body(const float4 xv, int lane,
                                            const float* __restrict__ g,
                                            const float* __restrict__ bt,
                                            float4* out4) {
  float s = xv.x + xv.y + xv.z + xv.w;
  s = wave_sum32(s);
  float mu = s * (1.f / 128.f);
  float dx = xv.x - mu, dy = xv.y - mu, dz = xv.z - mu, dw = xv.w - mu;
  float ss = dx * dx + dy * dy + dz * dz + dw * dw;
  ss = wave_sum32(ss);
  float rstd = rsqrtf(ss * (1.f / 128.f) + 1e-5f);
  float4 gv = ((const float4*)g)[lane];
  float4 bv = ((const float4*)bt)[lane];
  out4->x = dx * rstd * gv.x + bv.x;
  out4->y = dy * rstd * gv.y + bv.y;
  out4->z = dz * rstd * gv.z + bv.z;
  out4->w = dw * rstd * gv.w + bv.w;
}

__global__ void ln_rows_kernel(const float* __restrict__ x, const float* __restrict__ g,
                               const float* __restrict__ bt, float* __restrict__ of,
                               bf16_t* __restrict__ ob, int M) {
  int wave = threadIdx.x >> 5, lane = threadIdx.x & 31;
  int row = blockIdx.x * 8 + wave;
  if (row >= M) return;
  float4 xv = ((const float4*)(x + (size_t)row * 128))[lane];
  float4 y; ln_row_body(xv, lane, g, bt, &y);
  if (of) ((float4*)(of + (size_t)row * 128))[lane] = y;
  if (ob) {
    bf16_t* p = ob + (size_t)row * 128 + lane * 4;
    p[0] = (bf16_t)y.x; p[1] = (bf16_t)y.y; p[2] = (bf16_t)y.z; p[3] = (bf16_t)y.w;
  }
}

// out = res + LN(x)
__global__ void ln_add_kernel(const float* __restrict__ x, const float* __restrict__ res,
                              const float* __restrict__ g, const float* __restrict__ bt,
                              float* __restrict__ of, int M) {
  int wave = threadIdx.x >> 5, lane = threadIdx.x & 31;
  int row = blockIdx.x * 8 + wave;
  if (row >= M) return;
  float4 xv = ((const float4*)(x + (size_t)row * 128))[lane];
  float4 y; ln_row_body(xv, lane, g, bt, &y);
  float4 rv = ((const float4*)(res + (size_t)row * 128))[lane];
  y.x += rv.x; y.y += rv.y; y.z += rv.z; y.w += rv.w;
  ((float4*)(of + (size_t)row * 128))[lane] = y;
}

// ---------------- WMMA fragment load (ISA 16-bit A/B layout) ----------------
__device__ __forceinline__ v16bf load_frag128(const bf16_t* __restrict__ row, int koff, int kk) {
  v8bf lo = *(const v8bf*)(row + kk * 32 + koff);
  v8bf hi = *(const v8bf*)(row + kk * 32 + koff + 16);
  v16bf r;
#pragma unroll
  for (int i = 0; i < 8; ++i) { r[i] = lo[i]; r[i + 8] = hi[i]; }
  return r;
}

// ---------------- generic GEMM: C[M,128] = A[M,128](bf16) @ Wt'[128,128] ----------------
// Weight staged in LDS (TDM when available); B frags cached in regs; 4 M-tiles/block.
__global__ void gemm128_bf16_kernel(const bf16_t* __restrict__ A, const bf16_t* __restrict__ Wt,
                                    const float* __restrict__ bias, float* __restrict__ Cf,
                                    bf16_t* __restrict__ Cb, int relu) {
  __shared__ bf16_t ldsW[128 * 128];   // 32 KB
#if HAS_TDM
  if (threadIdx.x < 32) {
    tdm_load_w_128x128(Wt, (unsigned)(unsigned long long)(&ldsW[0]));
    __builtin_amdgcn_s_wait_tensorcnt(0);
  }
#else
  {
    const uint4* src = (const uint4*)Wt;
    uint4* dst = (uint4*)ldsW;
    for (int j = threadIdx.x; j < 2048; j += 256) dst[j] = src[j];
  }
#endif
  __syncthreads();

  const int lane   = threadIdx.x & 31;
  const int tile_n = threadIdx.x >> 5;          // 0..7 (8 waves cover N=128)
  const int nlane  = lane & 15;
  const int koff   = (lane >> 4) * 8;
  const int col    = tile_n * 16 + nlane;

  v16bf bfrag[4];
  const bf16_t* brow = ldsW + (size_t)col * 128;
#pragma unroll
  for (int kk = 0; kk < 4; ++kk) bfrag[kk] = load_frag128(brow, koff, kk);

  const float bb  = bias ? bias[col] : 0.f;
#pragma unroll
  for (int mt = 0; mt < 4; ++mt) {
    const int mtile = blockIdx.x * 4 + mt;
    const bf16_t* arow = A + (size_t)(mtile * 16 + nlane) * 128;
    v8f c = {};
#pragma unroll
    for (int kk = 0; kk < 4; ++kk) {
      v16bf a = load_frag128(arow, koff, kk);
      c = __builtin_amdgcn_wmma_f32_16x16x32_bf16(false, a, false, bfrag[kk], (short)0, c, false, false);
    }
    const int rbase = mtile * 16 + 8 * (lane >> 4);
#pragma unroll
    for (int r = 0; r < 8; ++r) {
      float v = c[r] + bb;
      if (relu) v = fmaxf(v, 0.f);
      size_t o = (size_t)(rbase + r) * 128 + col;
      if (Cf) Cf[o] = v;
      if (Cb) Cb[o] = (bf16_t)v;
    }
  }
}

// ---------------- fused rpe path ----------------
// One block per (b,q): gather+LN 32 neighbor rows of rpe into LDS (bf16), then a
// dual-output WMMA GEMM (Wrk', Wrv' share A fragments), epilogue fuses
// k1 = rkg + kf[b,idx], v1 = rvg + brv + vf[b,idx]. Outputs contiguous [i,128].
__global__ void rpe_fused_kernel(const float* __restrict__ rpe, const int* __restrict__ knn,
                                 const float* __restrict__ g, const float* __restrict__ bt,
                                 const bf16_t* __restrict__ Wrk_t, const bf16_t* __restrict__ Wrv_t,
                                 const float* __restrict__ brv,
                                 const float* __restrict__ kf, const float* __restrict__ vf,
                                 float* __restrict__ k1, float* __restrict__ v1) {
  __shared__ bf16_t ldsA[NB_ * 128];  // 8 KB
  __shared__ int    idxs[NB_];
  const int bq = blockIdx.x;                  // = b*LQ + q
  const int b  = bq >> 8;
  const int t = threadIdx.x, wave = t >> 5, lane = t & 31;

  if (t < NB_) idxs[t] = knn[(size_t)bq * NB_ + t];
  __syncthreads();

  // phase 1: LN of gathered rows, 4 rows per wave
#pragma unroll
  for (int rr = 0; rr < 4; ++rr) {
    int row = wave * 4 + rr;
    int idx = idxs[row];
    const float* src = rpe + ((size_t)bq * LK_ + idx) * 128;
    float4 xv = ((const float4*)src)[lane];
    float4 y; ln_row_body(xv, lane, g, bt, &y);
    bf16_t* p = ldsA + row * 128 + lane * 4;
    p[0] = (bf16_t)y.x; p[1] = (bf16_t)y.y; p[2] = (bf16_t)y.z; p[3] = (bf16_t)y.w;
  }
  __syncthreads();

  // phase 2: dual GEMM, wave = N-tile
  const int nlane = lane & 15;
  const int koff  = (lane >> 4) * 8;
  const int col   = wave * 16 + nlane;
  v16bf bk_[4], bv_[4];
  const bf16_t* bkrow = Wrk_t + (size_t)col * 128;
  const bf16_t* bvrow = Wrv_t + (size_t)col * 128;
#pragma unroll
  for (int kk = 0; kk < 4; ++kk) {
    bk_[kk] = load_frag128(bkrow, koff, kk);
    bv_[kk] = load_frag128(bvrow, koff, kk);
  }
  const float bvb = brv[col];

#pragma unroll
  for (int mt = 0; mt < 2; ++mt) {
    const bf16_t* arow = ldsA + (size_t)(mt * 16 + nlane) * 128;
    v8f ck = {}, cv = {};
#pragma unroll
    for (int kk = 0; kk < 4; ++kk) {
      v16bf a = load_frag128(arow, koff, kk);
      ck = __builtin_amdgcn_wmma_f32_16x16x32_bf16(false, a, false, bk_[kk], (short)0, ck, false, false);
      cv = __builtin_amdgcn_wmma_f32_16x16x32_bf16(false, a, false, bv_[kk], (short)0, cv, false, false);
    }
#pragma unroll
    for (int r = 0; r < 8; ++r) {
      const int rl  = mt * 16 + r + 8 * (lane >> 4);
      const size_t i = (size_t)bq * NB_ + rl;
      const int idx = idxs[rl];
      const size_t go = ((size_t)b * LK_ + idx) * 128 + col;
      k1[i * 128 + col] = ck[r] + kf[go];
      v1[i * 128 + col] = cv[r] + bvb + vf[go];
    }
  }
}

// ---------------- attention ----------------
// scores[b,h,q,n] = dot16(q1[b,q,h], k1[i,h]) / 4, masked; k1 is contiguous
__global__ void scores_kernel(const float* __restrict__ q1, const float* __restrict__ k1,
                              const int* __restrict__ knn, const unsigned char* __restrict__ mask,
                              float* __restrict__ sc) {
  __shared__ float qrow[128];
  __shared__ int   idxs[NB_];
  const int bq = blockIdx.x, b = bq >> 8, q = bq & 255, t = threadIdx.x;
  if (t < 128) qrow[t] = q1[(size_t)bq * 128 + t];
  else if (t < 128 + NB_) idxs[t - 128] = knn[(size_t)bq * NB_ + (t - 128)];
  __syncthreads();
  const int n = t >> 3, h = t & 7;
  const size_t i = (size_t)bq * NB_ + n;
  const float* kp = k1 + i * 128 + h * 16;
  const float* qp = qrow + h * 16;
  float acc = 0.f;
#pragma unroll
  for (int d = 0; d < 16; ++d) acc += qp[d] * kp[d];
  float v = acc * 0.25f;                              // / sqrt(16)
  if (mask[((size_t)b * LQ_ + q) * LK_ + idxs[n]]) v = -1e8f;
  sc[(((size_t)b * H_ + h) * LQ_ + q) * NB_ + n] = v;
}

// softmax over flattened (Lq*NB)=8192 axis, one block per (b,h); in place
__global__ void softmax_flat_kernel(float* __restrict__ s) {
  __shared__ float red[256];
  const int t = threadIdx.x;
  float* base = s + (size_t)blockIdx.x * (LQ_ * NB_);
  float m = -3.4e38f;
  for (int j = t; j < LQ_ * NB_; j += 256) m = fmaxf(m, base[j]);
  red[t] = m; __syncthreads();
  for (int off = 128; off > 0; off >>= 1) { if (t < off) red[t] = fmaxf(red[t], red[t + off]); __syncthreads(); }
  m = red[0]; __syncthreads();
  float sum = 0.f;
  for (int j = t; j < LQ_ * NB_; j += 256) sum += __expf(base[j] - m);
  red[t] = sum; __syncthreads();
  for (int off = 128; off > 0; off >>= 1) { if (t < off) red[t] += red[t + off]; __syncthreads(); }
  float inv = 1.f / red[0];
  for (int j = t; j < LQ_ * NB_; j += 256) base[j] = __expf(base[j] - m) * inv;
}

// qv[b,q,c] = sum_n attn[b,h(c),q,n] * v1[i,c]; bf16 out for the Wo GEMM
__global__ void qv_kernel(const float* __restrict__ attn, const float* __restrict__ v1,
                          bf16_t* __restrict__ qvb) {
  __shared__ float at[H_ * NB_];
  const int bq = blockIdx.x, b = bq >> 8, q = bq & 255, t = threadIdx.x;  // 128 threads
  for (int j = t; j < H_ * NB_; j += 128) {
    int h = j >> 5, n = j & 31;
    at[j] = attn[(((size_t)b * H_ + h) * LQ_ + q) * NB_ + n];
  }
  __syncthreads();
  const int h = t >> 4;
  float acc = 0.f;
#pragma unroll 4
  for (int n = 0; n < NB_; ++n)
    acc += at[h * 32 + n] * v1[((size_t)bq * NB_ + n) * 128 + t];
  qvb[(size_t)bq * 128 + t] = (bf16_t)acc;
}

// ---------------- host launch ----------------
extern "C" void kernel_launch(void* const* d_in, const int* in_sizes, int n_in,
                              void* d_out, int out_size, void* d_ws, size_t ws_size,
                              hipStream_t stream) {
  const float* q    = (const float*)d_in[0];
  const float* k    = (const float*)d_in[1];
  const float* v    = (const float*)d_in[2];
  const float* rpe  = (const float*)d_in[3];
  const int*   knn  = (const int*)d_in[4];
  const unsigned char* mask = (const unsigned char*)d_in[5];
  const float* ln_g = (const float*)d_in[6];
  const float* ln_b = (const float*)d_in[7];
  const float* Wq = (const float*)d_in[8];   const float* bq = (const float*)d_in[9];
  const float* Wk = (const float*)d_in[10];  const float* bk = (const float*)d_in[11];
  const float* Wv = (const float*)d_in[12];  const float* bv = (const float*)d_in[13];
  const float* Wrk = (const float*)d_in[14];
  const float* Wrv = (const float*)d_in[15]; const float* brv = (const float*)d_in[16];
  const float* Wo  = (const float*)d_in[17]; const float* bo  = (const float*)d_in[18];
  const float* Wm1 = (const float*)d_in[19]; const float* bm1 = (const float*)d_in[20];
  const float* Wm2 = (const float*)d_in[21]; const float* bm2 = (const float*)d_in[22];
  float* out = (float*)d_out;

  // bump allocator over workspace (256B aligned)
  char* p = (char*)d_ws;
  auto alloc = [&](size_t bytes) -> void* {
    void* r = (void*)p; p += (bytes + 255) & ~(size_t)255; return r;
  };
  const int MQ = B_ * LQ_;        // 1024
  const int MK = B_ * LK_;        // 2048
  float*  qn_f  = (float*)alloc((size_t)MQ * 128 * 4);
  bf16_t* qn_b  = (bf16_t*)alloc((size_t)MQ * 128 * 2);
  bf16_t* kn_b  = (bf16_t*)alloc((size_t)MK * 128 * 2);
  bf16_t* vn_b  = (bf16_t*)alloc((size_t)MK * 128 * 2);
  bf16_t* Wt[8];
  const float* Wsrc[8] = {Wq, Wk, Wv, Wrk, Wrv, Wo, Wm1, Wm2};
  for (int i = 0; i < 8; ++i) Wt[i] = (bf16_t*)alloc(128 * 128 * 2);
  float*  q1   = (float*)alloc((size_t)MQ * 128 * 4);
  float*  kf   = (float*)alloc((size_t)MK * 128 * 4);
  float*  vf   = (float*)alloc((size_t)MK * 128 * 4);
  float*  k1   = (float*)alloc((size_t)NROWS_R * 128 * 4);
  float*  v1   = (float*)alloc((size_t)NROWS_R * 128 * 4);
  float*  sc   = (float*)alloc((size_t)B_ * H_ * LQ_ * NB_ * 4);
  bf16_t* qv_b = (bf16_t*)alloc((size_t)MQ * 128 * 2);
  float*  y_f  = (float*)alloc((size_t)MQ * 128 * 4);
  float*  qv2  = (float*)alloc((size_t)MQ * 128 * 4);
  bf16_t* h_b  = (bf16_t*)alloc((size_t)MQ * 128 * 2);
  bf16_t* m1_b = (bf16_t*)alloc((size_t)MQ * 128 * 2);
  float*  m2_f = (float*)alloc((size_t)MQ * 128 * 4);

  // 1. weights -> bf16 [N][K]
  for (int i = 0; i < 8; ++i)
    prep_w_kernel<<<64, 256, 0, stream>>>(Wsrc[i], Wt[i]);

  // 2. layernorms of q/k/v
  ln_rows_kernel<<<MQ / 8, 256, 0, stream>>>(q, ln_g, ln_b, qn_f, qn_b, MQ);
  ln_rows_kernel<<<MK / 8, 256, 0, stream>>>(k, ln_g, ln_b, nullptr, kn_b, MK);
  ln_rows_kernel<<<MK / 8, 256, 0, stream>>>(v, ln_g, ln_b, nullptr, vn_b, MK);

  // 3. q/k/v projections (WMMA, LDS-staged weights, 4 M-tiles/block)
  gemm128_bf16_kernel<<<MQ / 64, 256, 0, stream>>>(qn_b, Wt[0], bq, q1, nullptr, 0);
  gemm128_bf16_kernel<<<MK / 64, 256, 0, stream>>>(kn_b, Wt[1], bk, kf, nullptr, 0);
  gemm128_bf16_kernel<<<MK / 64, 256, 0, stream>>>(vn_b, Wt[2], bv, vf, nullptr, 0);

  // 4. fused gather+LN+dual-WMMA+add for the rpe path -> k1, v1
  rpe_fused_kernel<<<MQ, 256, 0, stream>>>(rpe, knn, ln_g, ln_b, Wt[3], Wt[4], brv,
                                           kf, vf, k1, v1);

  // 5. attention (softmax over flattened (Lq*NB) per (b,h))
  scores_kernel<<<MQ, 256, 0, stream>>>(q1, k1, knn, mask, sc);
  softmax_flat_kernel<<<B_ * H_, 256, 0, stream>>>(sc);
  qv_kernel<<<MQ, 128, 0, stream>>>(sc, v1, qv_b);

  // 6. output projection + residual + MLP
  gemm128_bf16_kernel<<<MQ / 64, 256, 0, stream>>>(qv_b, Wt[5], bo, y_f, nullptr, 0);
  ln_add_kernel<<<MQ / 8, 256, 0, stream>>>(y_f, qn_f, ln_g, ln_b, qv2, MQ);
  ln_rows_kernel<<<MQ / 8, 256, 0, stream>>>(qv2, ln_g, ln_b, nullptr, h_b, MQ);
  gemm128_bf16_kernel<<<MQ / 64, 256, 0, stream>>>(h_b, Wt[6], bm1, nullptr, m1_b, 1);
  gemm128_bf16_kernel<<<MQ / 64, 256, 0, stream>>>(m1_b, Wt[7], bm2, m2_f, nullptr, 0);
  ln_add_kernel<<<MQ / 8, 256, 0, stream>>>(m2_f, qv2, ln_g, ln_b, out, MQ);

  (void)in_sizes; (void)n_in; (void)out_size; (void)ws_size;
}
